// CoWindowAttention_71786083385547
// MI455X (gfx1250) — matrix operand order repaired and checked by reading, not compiled, not run on hardware
//
#include <hip/hip_runtime.h>

typedef __attribute__((ext_vector_type(16))) __bf16 v16bf;
typedef __attribute__((ext_vector_type(8)))  float  v8f;

#define TOK 64
#define CH  128
#define NH  4
#define HD  32

// ---- WMMA fragment loaders (CDNA5 16-bit layouts, wave32) ----
// A 16x32 (MxK): lane holds row m=lane%16; half=lane/16:
//   VGPR0..3 -> K = 8*half + {0..7}, VGPR4..7 -> K = 16 + 8*half + {0..7}
__device__ __forceinline__ v16bf ld_fragA(const __bf16* row, int half) {
  union { v16bf v; uint4 u[2]; } t;
  t.u[0] = *(const uint4*)(row + half * 8);
  t.u[1] = *(const uint4*)(row + 16 + half * 8);
  return t.v;
}
// B 32x16 (KxN): lane holds col n=lane%16; half=lane/16 holds K = 8*half + {0..7}, 16 + 8*half + {0..7}
__device__ __forceinline__ v16bf ld_fragB(const __bf16* row, int half) {
  union { v16bf v; uint4 u[2]; } t;
  t.u[0] = *(const uint4*)(row + half * 8);
  t.u[1] = *(const uint4*)(row + 16 + half * 8);
  return t.v;
}

// One N-strip (16 cols) x 64 rows (4 M-tiles), K = KITERS*32.
// B fragment loaded once per K-step, reused for 4 M-tiles.
template <int KITERS>
__device__ __forceinline__ void strip_gemm(const __bf16* Abase, int aStride,
                                           const __bf16* Brow, int kBase,
                                           int l16, int half,
                                           v8f& c0, v8f& c1, v8f& c2, v8f& c3) {
#pragma unroll
  for (int kb = 0; kb < KITERS; ++kb) {
    const int ko = kBase + kb * 32;
    v16bf bf = ld_fragB(Brow + ko, half);
    v16bf a0 = ld_fragA(Abase + (l16     ) * aStride + ko, half);
    v16bf a1 = ld_fragA(Abase + (l16 + 16) * aStride + ko, half);
    v16bf a2 = ld_fragA(Abase + (l16 + 32) * aStride + ko, half);
    v16bf a3 = ld_fragA(Abase + (l16 + 48) * aStride + ko, half);
    c0 = __builtin_amdgcn_wmma_f32_16x16x32_bf16(false, a0, false, bf, (short)0, c0, false, false);
    c1 = __builtin_amdgcn_wmma_f32_16x16x32_bf16(false, a1, false, bf, (short)0, c1, false, false);
    c2 = __builtin_amdgcn_wmma_f32_16x16x32_bf16(false, a2, false, bf, (short)0, c2, false, false);
    c3 = __builtin_amdgcn_wmma_f32_16x16x32_bf16(false, a3, false, bf, (short)0, c3, false, false);
  }
}

// Setup kernel: convert weights fp32->bf16 AND materialize the relative-position
// bias table bias[h][i][j] = rpb[rpi[i*64+j]*NH + h] (block-invariant, 64KB, L2-resident).
// ws layout: [0, 65536) bf16 weights (qw|kvw|pw), then 16384 fp32 bias at byte 131072.
__global__ void setup_ws_kernel(const float* __restrict__ qw,
                                const float* __restrict__ kvw,
                                const float* __restrict__ pw,
                                const float* __restrict__ rpb,
                                const int*   __restrict__ rpi,
                                __bf16* __restrict__ wsb,
                                float*  __restrict__ biasT) {
  int i = blockIdx.x * 256 + threadIdx.x;           // 81920 total
  if (i < 65536) {
    float v;
    if (i < 16384)      v = qw[i];
    else if (i < 49152) v = kvw[i - 16384];
    else                v = pw[i - 49152];
    wsb[i] = (__bf16)v;
  } else {
    int t  = i - 65536;                             // 0..16383
    int h  = t >> 12;                               // 0..3
    int ij = t & 4095;                              // 0..4095
    biasT[h * 4096 + ij] = rpb[rpi[ij] * NH + h];
  }
}

// LDS layout (bytes):
//   Q  [64][128] bf16 @0       KK [64][128] bf16 @16384
//   VT [128][64] bf16 @32768   O  [64][128] bf16 @49152
//   P  [4][64][64] bf16 @65536
//   SB [4][64][64] f32 @98304 (aliased early by XA@98304, XS@114688)
// total 163840 B -> 2 blocks per 320KB WGP
__global__ __launch_bounds__(256)
void cowin_attn_kernel(const float* __restrict__ x,
                       const float* __restrict__ xs,
                       const float* __restrict__ maskp,
                       const float* __restrict__ biasT,
                       const float* __restrict__ qb,
                       const float* __restrict__ kvb,
                       const float* __restrict__ pb,
                       const __bf16* __restrict__ qwB,
                       const __bf16* __restrict__ kvwB,
                       const __bf16* __restrict__ pwB,
                       float* __restrict__ outp,
                       float* __restrict__ attnp) {
  extern __shared__ char smem[];
  __bf16* Q  = (__bf16*)(smem + 0);
  __bf16* KK = (__bf16*)(smem + 16384);
  __bf16* VT = (__bf16*)(smem + 32768);
  __bf16* O  = (__bf16*)(smem + 49152);
  __bf16* P  = (__bf16*)(smem + 65536);
  float*  SB = (float*)(smem + 98304);
  __bf16* XA = (__bf16*)(smem + 98304);
  __bf16* XS = (__bf16*)(smem + 114688);

  const int tid  = threadIdx.x;
  const int lane = tid & 31;
  const int wv   = tid >> 5;        // wave 0..7
  const int half = lane >> 4;
  const int l16  = lane & 15;
  const int b    = blockIdx.x;
  const float scale = 0.17677669529663687f;   // 32^-0.5

  // ---- phase 0: load x / x_selected, convert fp32 -> bf16 into LDS ----
  {
    const float4* xg  = (const float4*)(x  + (size_t)b * TOK * CH);
    const float4* xsg = (const float4*)(xs + (size_t)b * TOK * CH);
#pragma unroll
    for (int it = 0; it < 8; ++it) {
      int idx = it * 256 + tid;               // 2048 float4 = 64x128
      float4 a = xg[idx];
      float4 s = xsg[idx];
      union { __bf16 h[4]; uint2 u; } pa, ps;
      pa.h[0] = (__bf16)a.x; pa.h[1] = (__bf16)a.y; pa.h[2] = (__bf16)a.z; pa.h[3] = (__bf16)a.w;
      ps.h[0] = (__bf16)s.x; ps.h[1] = (__bf16)s.y; ps.h[2] = (__bf16)s.z; ps.h[3] = (__bf16)s.w;
      *(uint2*)(XA + idx * 4) = pa.u;
      *(uint2*)(XS + idx * 4) = ps.u;
    }
  }
  __syncthreads();

  // ---- phase 1a: q = x @ qw^T + qb, scaled; wave owns N-strip nt = wv ----
  {
    const int ng = wv * 16 + l16;             // 0..127
    v8f c0 = {}, c1 = {}, c2 = {}, c3 = {};
    strip_gemm<4>(XA, CH, qwB + ng * CH, 0, l16, half, c0, c1, c2, c3);
    const float bias = qb[ng];
#pragma unroll
    for (int g = 0; g < 8; ++g) {
      int ml = half * 8 + g;
      Q[(ml     ) * CH + ng] = (__bf16)((c0[g] + bias) * scale);
      Q[(ml + 16) * CH + ng] = (__bf16)((c1[g] + bias) * scale);
      Q[(ml + 32) * CH + ng] = (__bf16)((c2[g] + bias) * scale);
      Q[(ml + 48) * CH + ng] = (__bf16)((c3[g] + bias) * scale);
    }
  }
  // ---- phase 1b: kv = xs @ kvw^T + kvb; 16 strips, 2 per wave; v stored transposed ----
#pragma unroll
  for (int ss = 0; ss < 2; ++ss) {
    const int nt = wv * 2 + ss;               // 0..15
    const int ng = nt * 16 + l16;             // 0..255
    v8f c0 = {}, c1 = {}, c2 = {}, c3 = {};
    strip_gemm<4>(XS, CH, kvwB + ng * CH, 0, l16, half, c0, c1, c2, c3);
    const float bias = kvb[ng];
#pragma unroll
    for (int g = 0; g < 8; ++g) {
      int ml = half * 8 + g;
      float v0 = c0[g] + bias, v1 = c1[g] + bias, v2 = c2[g] + bias, v3 = c3[g] + bias;
      if (ng < 128) {                          // K part, row-major [tok][128]
        KK[(ml     ) * CH + ng] = (__bf16)v0;
        KK[(ml + 16) * CH + ng] = (__bf16)v1;
        KK[(ml + 32) * CH + ng] = (__bf16)v2;
        KK[(ml + 48) * CH + ng] = (__bf16)v3;
      } else {                                 // V part, transposed [d][tok]
        int d = ng - 128;
        VT[d * TOK + (ml     )] = (__bf16)v0;
        VT[d * TOK + (ml + 16)] = (__bf16)v1;
        VT[d * TOK + (ml + 32)] = (__bf16)v2;
        VT[d * TOK + (ml + 48)] = (__bf16)v3;
      }
    }
  }
  __syncthreads();

  // ---- phase 2: scores S = q_h @ k_h^T + bias + mask ; 16 strips (h,nt), 2/wave ----
  {
    const int wIdx = b & 63;
    const float* mbase = maskp + (size_t)wIdx * 64 * 64;
#pragma unroll
    for (int ss = 0; ss < 2; ++ss) {
      const int s  = wv * 2 + ss;             // 0..15
      const int h  = s >> 2;
      const int nt = s & 3;
      const int jg = nt * 16 + l16;           // key index j
      v8f c0 = {}, c1 = {}, c2 = {}, c3 = {};
      strip_gemm<1>(Q, CH, KK + jg * CH, h * 32, l16, half, c0, c1, c2, c3);
      float* srow = SB + (h * 64) * 64;
      const float* brow = biasT + (h * 64) * 64;   // precomputed rpb gather
#pragma unroll
      for (int g = 0; g < 8; ++g) {
        int ml = half * 8 + g;
        int i0 = ml, i1 = ml + 16, i2 = ml + 32, i3 = ml + 48;
        srow[i0 * 64 + jg] = c0[g] + brow[i0 * 64 + jg] + mbase[i0 * 64 + jg];
        srow[i1 * 64 + jg] = c1[g] + brow[i1 * 64 + jg] + mbase[i1 * 64 + jg];
        srow[i2 * 64 + jg] = c2[g] + brow[i2 * 64 + jg] + mbase[i2 * 64 + jg];
        srow[i3 * 64 + jg] = c3[g] + brow[i3 * 64 + jg] + mbase[i3 * 64 + jg];
      }
    }
  }
  __syncthreads();

  // ---- phase 3: softmax, one thread per (h,i) row; write bf16 probs + fp32 norm ----
  {
    const int h = tid >> 6, i = tid & 63;
    float* row = SB + (h * 64 + i) * 64;
    __bf16* prow = P + (h * 64 + i) * 64;
    float mx = -3.4e38f;
    for (int jj = 0; jj < 64; ++jj) { int j = (jj + lane) & 63; mx = fmaxf(mx, row[j]); }
    float sum = 0.f;
    for (int jj = 0; jj < 64; ++jj) {
      int j = (jj + lane) & 63;
      float e = __expf(row[j] - mx);
      row[j] = e; sum += e;
    }
    float inv = 1.f / sum;
    for (int jj = 0; jj < 64; ++jj) {
      int j = (jj + lane) & 63;
      float p = row[j] * inv;
      row[j] = p;
      prow[j] = (__bf16)p;
    }
  }
  __syncthreads();

  // ---- phase 4a: coalesced store of attn (fp32) ----
  {
    float4* ag = (float4*)(attnp + (size_t)b * (NH * 64 * 64));
    const float4* sb4 = (const float4*)SB;
#pragma unroll
    for (int it = 0; it < 16; ++it) ag[it * 256 + tid] = sb4[it * 256 + tid];
  }
  // ---- phase 4b: O_h = P_h @ V_h ; 8 strips (h, half-of-head), 1/wave ----
  {
    const int h = wv >> 1, ntv = wv & 1;
    const int dg = 32 * h + ntv * 16 + l16;   // output channel
    v8f c0 = {}, c1 = {}, c2 = {}, c3 = {};
    strip_gemm<2>(P + h * 64 * 64, 64, VT + dg * TOK, 0, l16, half, c0, c1, c2, c3);
#pragma unroll
    for (int g = 0; g < 8; ++g) {
      int ml = half * 8 + g;
      O[(ml     ) * CH + dg] = (__bf16)c0[g];
      O[(ml + 16) * CH + dg] = (__bf16)c1[g];
      O[(ml + 32) * CH + dg] = (__bf16)c2[g];
      O[(ml + 48) * CH + dg] = (__bf16)c3[g];
    }
  }
  __syncthreads();

  // ---- phase 5: out = O @ pw^T + pb -> stage fp32 in SB, coalesced store ----
  {
    const int ng = wv * 16 + l16;
    v8f c0 = {}, c1 = {}, c2 = {}, c3 = {};
    strip_gemm<4>(O, CH, pwB + ng * CH, 0, l16, half, c0, c1, c2, c3);
    const float bias = pb[ng];
#pragma unroll
    for (int g = 0; g < 8; ++g) {
      int ml = half * 8 + g;
      SB[(ml     ) * CH + ng] = c0[g] + bias;
      SB[(ml + 16) * CH + ng] = c1[g] + bias;
      SB[(ml + 32) * CH + ng] = c2[g] + bias;
      SB[(ml + 48) * CH + ng] = c3[g] + bias;
    }
  }
  __syncthreads();
  {
    float4* og = (float4*)(outp + (size_t)b * TOK * CH);
    const float4* sb4 = (const float4*)SB;
#pragma unroll
    for (int it = 0; it < 8; ++it) og[it * 256 + tid] = sb4[it * 256 + tid];
  }
}

extern "C" void kernel_launch(void* const* d_in, const int* in_sizes, int n_in,
                              void* d_out, int out_size, void* d_ws, size_t ws_size,
                              hipStream_t stream) {
  const float* x    = (const float*)d_in[0];
  const float* xs   = (const float*)d_in[1];
  const float* mask = (const float*)d_in[2];
  const float* rpb  = (const float*)d_in[3];
  const int*   rpi  = (const int*)d_in[4];
  const float* qw   = (const float*)d_in[5];
  const float* qb   = (const float*)d_in[6];
  const float* kvw  = (const float*)d_in[7];
  const float* kvb  = (const float*)d_in[8];
  const float* pw   = (const float*)d_in[9];
  const float* pb   = (const float*)d_in[10];

  const int B = in_sizes[0] / (TOK * CH);     // 4096 windows

  // ws: 65536 bf16 weights (128KB) then 16384 fp32 bias (64KB)
  __bf16* wsb   = (__bf16*)d_ws;
  float*  biasT = (float*)((char*)d_ws + 131072);
  setup_ws_kernel<<<320, 256, 0, stream>>>(qw, kvw, pw, rpb, rpi, wsb, biasT);

  float* outp  = (float*)d_out;
  float* attnp = outp + (size_t)B * TOK * CH;
  const __bf16* qwB  = wsb;
  const __bf16* kvwB = wsb + 16384;
  const __bf16* pwB  = wsb + 49152;

  cowin_attn_kernel<<<B, 256, 163840, stream>>>(x, xs, mask, biasT,
                                                qb, kvb, pb, qwB, kvwB, pwB,
                                                outp, attnp);
}
